// EducationalLLM_18528488915389
// MI455X (gfx1250) — matrix operand order repaired and checked by reading, not compiled, not run on hardware
//
#include <hip/hip_runtime.h>
#include <hip/hip_bf16.h>
#include <math.h>

// ---------------------------------------------------------------------------
// GPT-2-ish forward (L=4, D=1024, H=16, DK=64, S=2048, DFF=4096, V=50257)
// Heavy matmuls on v_wmma_f32_16x16x32_f16 (wave32 WMMA), fp32->f16 staged
// through double-buffered LDS with vectorized (b128) global loads.
// ---------------------------------------------------------------------------

#define VOCAB 50257
#define DMODEL 1024
#define NLAYER 4
#define NHEAD 16
#define SEQ 2048
#define DFFN 4096
#define DKH (DMODEL / NHEAD)   // 64
#define LN_EPS 1e-5f

typedef __attribute__((ext_vector_type(16))) _Float16 v16h;
typedef __attribute__((ext_vector_type(8)))  _Float16 v8h;
typedef __attribute__((ext_vector_type(4)))  _Float16 v4h;
typedef __attribute__((ext_vector_type(8)))  float    v8f;
typedef __attribute__((ext_vector_type(4)))  float    v4f;

// Tile config: 128x128 tile, K-step 32, 256 threads = 8 waves.
// Each wave computes a 16x128 strip (8 accumulators -> 8 WMMA per k-step).
#define BM 128
#define BN 128
#define BK 32

// BMODE: 0 = B row-major [K,N], vector loads along N (ldb % 4 == 0)
//        1 = B row-major [K,N], scalar guarded loads (lm_head, ldb=50257)
//        2 = B^T: B given as [N,K] row-major, vector loads along K (scores)
// act:   0 = none, 1 = exact GELU, 2 = scale + causal mask
template <int BMODE>
__global__ __launch_bounds__(256)
void gemm_wmma_kernel(const float* __restrict__ A,
                      const float* __restrict__ B,
                      const float* __restrict__ bias,      // nullable
                      const float* __restrict__ residual,  // nullable, ld = ldc
                      float* __restrict__ C,
                      int M, int N, int K,
                      int lda, int ldb, int ldc,
                      long long aBatch, long long bBatch, long long cBatch,
                      int act, float scale)
{
    __shared__ __align__(64) _Float16 As[2][BM][BK];   // [m][k]
    __shared__ __align__(64) _Float16 Bs[2][BN][BK];   // [n][k] (K-transposed)

    A += (long long)blockIdx.z * aBatch;
    B += (long long)blockIdx.z * bBatch;
    C += (long long)blockIdx.z * cBatch;

    const int tid  = threadIdx.x;
    const int lane = tid & 31;
    const int wave = tid >> 5;          // 0..7 -> row strip m0 = wave*16
    const int hi   = lane >> 4;         // lane half
    const int l16  = lane & 15;

    const int blockM = blockIdx.y * BM;
    const int blockN = blockIdx.x * BN;

    // ---- causal early-out: whole block above the diagonal ----
    if (act == 2 && blockN > blockM + (BM - 1)) {
        #pragma unroll
        for (int i = 0; i < 8; ++i) {
            int gn = blockN + i * 16 + l16;
            if (gn >= N) continue;
            #pragma unroll
            for (int r = 0; r < 8; ++r) {
                int gm = blockM + wave * 16 + hi * 8 + r;
                C[(long long)gm * ldc + gn] = -1e9f;
            }
        }
        return;
    }

    v8f acc[8];
    #pragma unroll
    for (int i = 0; i < 8; ++i) acc[i] = (v8f){};

    // ---- hoisted per-thread staging pointers ----
    // A mapping: m = (tid>>3) + it*32 (0..127), k4 = tid&7
    const int am  = tid >> 3;
    const int ak4 = tid & 7;
    const float* aPtr = A + (long long)(blockM + am) * lda + ak4 * 4;
    const long long aStep = 32LL * lda;          // 32 rows per it

    // B mode 0: kk = (tid>>5) + it*8, n4 = tid&31
    // B mode 2: n  = (tid>>3) + it*32, k4 = tid&7
    // B mode 1: n  = tid&127, kk = (tid>>7) + it*2
    const int b0kk = tid >> 5, b0n4 = tid & 31;
    const int b2n  = tid >> 3, b2k4 = tid & 7;
    const int b1n  = tid & 127, b1kk = tid >> 7;

    v4f ra[4];
    v4f rbv[4];
    float rbs[16];

    auto loadA = [&](int k0) {
        #pragma unroll
        for (int it = 0; it < 4; ++it)
            ra[it] = *(const v4f*)&aPtr[it * aStep + k0];
    };
    auto storeA = [&](int buf) {
        #pragma unroll
        for (int it = 0; it < 4; ++it)
            *(v4h*)&As[buf][am + it * 32][ak4 * 4] = __builtin_convertvector(ra[it], v4h);
    };

    auto loadB = [&](int k0) {
        if (BMODE == 0) {
            const int gn = blockN + b0n4 * 4;
            const float* bPtr = B + (long long)(k0 + b0kk) * ldb + gn;
            const long long bStep = 8LL * ldb;
            #pragma unroll
            for (int it = 0; it < 4; ++it) {
                v4f t = {};
                if (gn + 3 < N) {
                    t = *(const v4f*)&bPtr[it * bStep];
                } else {
                    #pragma unroll
                    for (int j = 0; j < 4; ++j)
                        if (gn + j < N) t[j] = bPtr[it * bStep + j];
                }
                rbv[it] = t;
            }
        } else if (BMODE == 2) {
            const float* bPtr = B + (long long)(blockN + b2n) * ldb + k0 + b2k4 * 4;
            const long long bStep = 32LL * ldb;
            #pragma unroll
            for (int it = 0; it < 4; ++it) {
                v4f t = {};
                if (blockN + b2n + it * 32 < N)
                    t = *(const v4f*)&bPtr[it * bStep];
                rbv[it] = t;
            }
        } else {
            const int gn = blockN + b1n;
            const float* bPtr = B + (long long)(k0 + b1kk) * ldb + gn;
            const long long bStep = 2LL * ldb;
            #pragma unroll
            for (int it = 0; it < 16; ++it)
                rbs[it] = (gn < N) ? bPtr[it * bStep] : 0.0f;
        }
    };
    auto storeB = [&](int buf) {
        if (BMODE == 0) {
            #pragma unroll
            for (int it = 0; it < 4; ++it) {
                int kk = b0kk + it * 8;
                v4h hh = __builtin_convertvector(rbv[it], v4h);
                #pragma unroll
                for (int j = 0; j < 4; ++j) Bs[buf][b0n4 * 4 + j][kk] = hh[j];
            }
        } else if (BMODE == 2) {
            #pragma unroll
            for (int it = 0; it < 4; ++it)
                *(v4h*)&Bs[buf][b2n + it * 32][b2k4 * 4] = __builtin_convertvector(rbv[it], v4h);
        } else {
            #pragma unroll
            for (int it = 0; it < 16; ++it)
                Bs[buf][b1n][b1kk + it * 2] = (_Float16)rbs[it];
        }
    };

    // ---- pipelined main loop: one barrier per k-step, double-buffered LDS ----
    loadA(0);
    loadB(0);
    int buf = 0;
    for (int k0 = 0; k0 < K; k0 += BK) {
        storeA(buf);
        storeB(buf);
        __syncthreads();

        if (k0 + BK < K) {          // issue next tile's global loads (no wait)
            loadA(k0 + BK);
            loadB(k0 + BK);
        }

        // Fetch A fragment + ALL 8 B fragments first, then issue 8 WMMAs
        // back-to-back so DS waits stagger instead of fully draining per MMA.
        v16h afrag;
        {
            afrag = *(const v16h*)&As[buf][wave * 16 + l16][hi * 16];
        }
        v16h bf[8];
        #pragma unroll
        for (int i = 0; i < 8; ++i)
            bf[i] = *(const v16h*)&Bs[buf][i * 16 + l16][hi * 16];

        #pragma unroll
        for (int i = 0; i < 8; ++i)
            acc[i] = __builtin_amdgcn_wmma_f32_16x16x32_f16(false, afrag, false, bf[i],
                                                            (short)0, acc[i], false, false);
        buf ^= 1;
    }

    // ---- epilogue: C layout VGPR r -> M = wave*16 + hi*8 + r, N = i*16 + l16 ----
    #pragma unroll
    for (int i = 0; i < 8; ++i) {
        int gn = blockN + i * 16 + l16;
        if (gn >= N) continue;
        float bv = bias ? bias[gn] : 0.0f;
        #pragma unroll
        for (int r = 0; r < 8; ++r) {
            int gm = blockM + wave * 16 + hi * 8 + r;
            float v = acc[i][r] + bv;
            if (act == 1) {
                v = 0.5f * v * (1.0f + erff(v * 0.70710678118654752f));
            } else if (act == 2) {
                v = (gn <= gm) ? v * scale : -1e9f;
            }
            if (residual) v += residual[(long long)gm * ldc + gn];
            C[(long long)gm * ldc + gn] = v;
        }
    }
}

// ---------------------------------------------------------------------------
// Embedding gather + positional add
// ---------------------------------------------------------------------------
__global__ void embed_kernel(const int* __restrict__ ids,
                             const float* __restrict__ tok,
                             const float* __restrict__ pos,
                             float* __restrict__ x)
{
    int t = blockIdx.x;
    long long tokrow = (long long)ids[t] * DMODEL;
    for (int d = threadIdx.x; d < DMODEL; d += blockDim.x) {
        x[(long long)t * DMODEL + d] = tok[tokrow + d] + pos[(long long)t * DMODEL + d];
    }
}

// ---------------------------------------------------------------------------
// LayerNorm over D=1024; one block (256 thr) per row.
// ---------------------------------------------------------------------------
__global__ __launch_bounds__(256)
void layernorm_kernel(const float* __restrict__ x,
                      const float* __restrict__ s,
                      const float* __restrict__ b,
                      float* __restrict__ y)
{
    __shared__ float red[256];
    const long long row = (long long)blockIdx.x * DMODEL;
    const int tid = threadIdx.x;

    float sum = 0.0f;
    float vals[DMODEL / 256];
    #pragma unroll
    for (int i = 0; i < DMODEL / 256; ++i) {
        vals[i] = x[row + tid + i * 256];
        sum += vals[i];
    }
    red[tid] = sum; __syncthreads();
    for (int off = 128; off > 0; off >>= 1) {
        if (tid < off) red[tid] += red[tid + off];
        __syncthreads();
    }
    float mu = red[0] * (1.0f / DMODEL);
    __syncthreads();

    float vsum = 0.0f;
    #pragma unroll
    for (int i = 0; i < DMODEL / 256; ++i) {
        float d = vals[i] - mu;
        vsum += d * d;
    }
    red[tid] = vsum; __syncthreads();
    for (int off = 128; off > 0; off >>= 1) {
        if (tid < off) red[tid] += red[tid + off];
        __syncthreads();
    }
    float rstd = rsqrtf(red[0] * (1.0f / DMODEL) + LN_EPS);

    #pragma unroll
    for (int i = 0; i < DMODEL / 256; ++i) {
        int d = tid + i * 256;
        y[row + d] = (vals[i] - mu) * rstd * s[d] + b[d];
    }
}

// ---------------------------------------------------------------------------
// Row softmax over SEQ=2048 keys; grid = (SEQ rows, NHEAD).
// ---------------------------------------------------------------------------
__global__ __launch_bounds__(256)
void softmax_kernel(float* __restrict__ scores)
{
    __shared__ float red[256];
    const int tid = threadIdx.x;
    float* row = scores + ((long long)blockIdx.y * SEQ + blockIdx.x) * SEQ;

    float vals[SEQ / 256];
    float mx = -3.4e38f;
    #pragma unroll
    for (int i = 0; i < SEQ / 256; ++i) {
        vals[i] = row[tid + i * 256];
        mx = fmaxf(mx, vals[i]);
    }
    red[tid] = mx; __syncthreads();
    for (int off = 128; off > 0; off >>= 1) {
        if (tid < off) red[tid] = fmaxf(red[tid], red[tid + off]);
        __syncthreads();
    }
    mx = red[0]; __syncthreads();

    float sum = 0.0f;
    #pragma unroll
    for (int i = 0; i < SEQ / 256; ++i) {
        vals[i] = __expf(vals[i] - mx);
        sum += vals[i];
    }
    red[tid] = sum; __syncthreads();
    for (int off = 128; off > 0; off >>= 1) {
        if (tid < off) red[tid] += red[tid + off];
        __syncthreads();
    }
    float inv = 1.0f / red[0];
    #pragma unroll
    for (int i = 0; i < SEQ / 256; ++i) {
        row[tid + i * 256] = vals[i] * inv;
    }
}

// ---------------------------------------------------------------------------
// Host-side orchestration
// ---------------------------------------------------------------------------
static inline void gemm_launch(const float* A, const float* B, const float* bias,
                               const float* residual, float* C,
                               int M, int N, int K, int lda, int ldb, int ldc,
                               long long aBatch, long long bBatch, long long cBatch,
                               int batch, int transB, int act, float scale,
                               hipStream_t stream)
{
    dim3 grid((N + BN - 1) / BN, M / BM, batch);
    dim3 block(256);
    if (transB) {
        gemm_wmma_kernel<2><<<grid, block, 0, stream>>>(A, B, bias, residual, C,
            M, N, K, lda, ldb, ldc, aBatch, bBatch, cBatch, act, scale);
    } else if ((ldb & 3) == 0) {
        gemm_wmma_kernel<0><<<grid, block, 0, stream>>>(A, B, bias, residual, C,
            M, N, K, lda, ldb, ldc, aBatch, bBatch, cBatch, act, scale);
    } else {
        gemm_wmma_kernel<1><<<grid, block, 0, stream>>>(A, B, bias, residual, C,
            M, N, K, lda, ldb, ldc, aBatch, bBatch, cBatch, act, scale);
    }
}

extern "C" void kernel_launch(void* const* d_in, const int* in_sizes, int n_in,
                              void* d_out, int out_size, void* d_ws, size_t ws_size,
                              hipStream_t stream)
{
    (void)in_sizes; (void)n_in; (void)out_size; (void)ws_size;

    const int*   ids   = (const int*)  d_in[0];
    const float* tok   = (const float*)d_in[1];
    const float* pos   = (const float*)d_in[2];
    const float* wq    = (const float*)d_in[3];
    const float* bq    = (const float*)d_in[4];
    const float* wk    = (const float*)d_in[5];
    const float* bk    = (const float*)d_in[6];
    const float* wv    = (const float*)d_in[7];
    const float* bv    = (const float*)d_in[8];
    const float* wo    = (const float*)d_in[9];
    const float* bo    = (const float*)d_in[10];
    const float* ln1s  = (const float*)d_in[11];
    const float* ln1b  = (const float*)d_in[12];
    const float* ln2s  = (const float*)d_in[13];
    const float* ln2b  = (const float*)d_in[14];
    const float* w1    = (const float*)d_in[15];
    const float* b1    = (const float*)d_in[16];
    const float* w2    = (const float*)d_in[17];
    const float* b2    = (const float*)d_in[18];
    const float* lnfs  = (const float*)d_in[19];
    const float* lnfb  = (const float*)d_in[20];
    const float* lmw   = (const float*)d_in[21];
    float* out = (float*)d_out;

    // ---- workspace carve-out (floats) ----
    const size_t SD = (size_t)SEQ * DMODEL;
    float* x      = (float*)d_ws;
    float* h      = x + SD;
    float* q      = h + SD;
    float* kbuf   = q + SD;
    float* vbuf   = kbuf + SD;
    float* ctx    = vbuf + SD;
    float* ff     = ctx + SD;                        // SEQ*DFFN
    float* scores = ff + (size_t)SEQ * DFFN;         // NHEAD*SEQ*SEQ

    embed_kernel<<<dim3(SEQ), dim3(256), 0, stream>>>(ids, tok, pos, x);

    const float attn_scale = 1.0f / 8.0f;            // 1/sqrt(DK)

    for (int l = 0; l < NLAYER; ++l) {
        const size_t oDD  = (size_t)l * DMODEL * DMODEL;
        const size_t oD   = (size_t)l * DMODEL;
        const size_t oDF  = (size_t)l * DMODEL * DFFN;
        const size_t oF   = (size_t)l * DFFN;

        layernorm_kernel<<<dim3(SEQ), dim3(256), 0, stream>>>(x, ln1s + oD, ln1b + oD, h);

        gemm_launch(h, wq + oDD, bq + oD, nullptr, q,    SEQ, DMODEL, DMODEL,
                    DMODEL, DMODEL, DMODEL, 0, 0, 0, 1, 0, 0, 0.f, stream);
        gemm_launch(h, wk + oDD, bk + oD, nullptr, kbuf, SEQ, DMODEL, DMODEL,
                    DMODEL, DMODEL, DMODEL, 0, 0, 0, 1, 0, 0, 0.f, stream);
        gemm_launch(h, wv + oDD, bv + oD, nullptr, vbuf, SEQ, DMODEL, DMODEL,
                    DMODEL, DMODEL, DMODEL, 0, 0, 0, 1, 0, 0, 0.f, stream);

        // scores[h] = causal_mask(scale * Q_h @ K_h^T)   (batched over heads)
        gemm_launch(q, kbuf, nullptr, nullptr, scores, SEQ, SEQ, DKH,
                    DMODEL, DMODEL, SEQ,
                    (long long)DKH, (long long)DKH, (long long)SEQ * SEQ,
                    NHEAD, /*transB=*/1, /*act=*/2, attn_scale, stream);

        softmax_kernel<<<dim3(SEQ, NHEAD), dim3(256), 0, stream>>>(scores);

        // ctx_h = attn_h @ V_h   (batched over heads)
        gemm_launch(scores, vbuf, nullptr, nullptr, ctx, SEQ, DKH, SEQ,
                    SEQ, DMODEL, DMODEL,
                    (long long)SEQ * SEQ, (long long)DKH, (long long)DKH,
                    NHEAD, 0, 0, 0.f, stream);

        // x = x + ctx @ Wo + bo
        gemm_launch(ctx, wo + oDD, bo + oD, /*residual=*/x, x, SEQ, DMODEL, DMODEL,
                    DMODEL, DMODEL, DMODEL, 0, 0, 0, 1, 0, 0, 0.f, stream);

        layernorm_kernel<<<dim3(SEQ), dim3(256), 0, stream>>>(x, ln2s + oD, ln2b + oD, h);

        // ff = GELU(h @ W1 + b1)
        gemm_launch(h, w1 + oDF, b1 + oF, nullptr, ff, SEQ, DFFN, DMODEL,
                    DMODEL, DFFN, DFFN, 0, 0, 0, 1, 0, /*act=*/1, 0.f, stream);

        // x = x + ff @ W2 + b2
        gemm_launch(ff, w2 + oDF, b2 + oD, /*residual=*/x, x, SEQ, DMODEL, DFFN,
                    DFFN, DMODEL, DMODEL, 0, 0, 0, 1, 0, 0, 0.f, stream);
    }

    layernorm_kernel<<<dim3(SEQ), dim3(256), 0, stream>>>(x, lnfs, lnfb, h);
    gemm_launch(h, lmw, nullptr, nullptr, out, SEQ, VOCAB, DMODEL,
                DMODEL, VOCAB, VOCAB, 0, 0, 0, 1, 0, 0, 0.f, stream);
}
